// MultiHeadAttention_549755814006
// MI455X (gfx1250) — matrix operand order repaired and checked by reading, not compiled, not run on hardware
//
#include <hip/hip_runtime.h>
#include <hip/hip_bf16.h>
#include <stdint.h>

// Shapes
#define BB 8
#define LL 1024
#define HH 8
#define DKK 80
#define DMM 640

typedef __attribute__((ext_vector_type(16))) _Float16 v16h;
typedef __attribute__((ext_vector_type(8)))  float    v8f;

union Frag16 {
    v16h v;
    unsigned int u[8];
    uint4 q4[2];
};

// A-matrix (16x32 f16) half-index -> K offset within a 32-wide chunk.
// VGPR j holds K = (j<4 ? 2j : 16+2(j-4)) + hi*8  (and K+1), hi = lane>>4.
__device__ __forceinline__ int kkoff(int j, int hi) {
    return (j < 4 ? 2 * j : 16 + 2 * (j - 4)) + hi * 8;
}

// ---------------------------------------------------------------------------
// Kernel 0: skdiv[b][h][key] = sum_d K[b,key,h,d]^2 / 13
// ---------------------------------------------------------------------------
__global__ void l2mha_sk_kernel(const float* __restrict__ k, float* __restrict__ skdiv) {
    int gid = blockIdx.x * blockDim.x + threadIdx.x;
    if (gid >= BB * HH * LL) return;
    int b   = gid / (HH * LL);
    int rem = gid % (HH * LL);
    int h   = rem / LL;
    int key = rem % LL;
    const float* kp = k + ((size_t)(b * LL + key)) * DMM + h * DKK;
    float s = 0.f;
#pragma unroll
    for (int i = 0; i < 20; ++i) {
        float4 x = *(const float4*)(kp + 4 * i);
        s += x.x * x.x + x.y * x.y + x.z * x.z + x.w * x.w;
    }
    skdiv[gid] = s * (1.0f / 13.0f);
}

// ---------------------------------------------------------------------------
// Kernel 0b: convert fc_w (640x640 f32) -> f16
// ---------------------------------------------------------------------------
__global__ void l2mha_cvt_kernel(const float* __restrict__ w, _Float16* __restrict__ w16, int n) {
    int gid = blockIdx.x * blockDim.x + threadIdx.x;
    if (gid < n) w16[gid] = (_Float16)w[gid];
}

// ---------------------------------------------------------------------------
// Kernel 1: flash attention with f16 WMMA, double-buffered LDS staging.
// Grid: B*H*(L/64) blocks of 128 threads (4 waves); wave w owns a 16-row
// q-tile.  Per 32-key iteration: issue next tile's global loads, run
// 6 QK^T WMMAs + online softmax + P transpose (per-wave LDS) + 5 PV WMMAs,
// then convert/store next tile into the other LDS buffer.  One barrier/iter.
// ---------------------------------------------------------------------------
__global__ __launch_bounds__(128) void l2mha_attn_kernel(
    const float* __restrict__ qg, const float* __restrict__ kg,
    const float* __restrict__ vg, const float* __restrict__ skdiv,
    _Float16* __restrict__ attn16)
{
    __shared__ __align__(16) _Float16 kbuf[2][32][104];   // keys x d(96+pad), f16
    __shared__ __align__(16) _Float16 vbufT[2][80][40];   // d x keys(32+pad), f16
    __shared__ __align__(16) _Float16 pbuf[4][16][36];    // per-wave P transpose

    const int tid  = threadIdx.x;
    const int wv   = tid >> 5;
    const int lane = tid & 31;
    const int n    = lane & 15;
    const int hi   = lane >> 4;

    const int bid = blockIdx.x;
    const int b  = bid >> 7;         // 1024 blocks = 8 * 8 * 16
    const int h  = (bid >> 4) & 7;
    const int qt = bid & 15;
    const int q0 = qt * 64 + wv * 16;

    // staging thread mapping
    const int keyl = tid >> 2;
    const int qtr  = tid & 3;
    const int d0   = qtr * 20;
    const size_t stbase = ((size_t)(b * LL + keyl)) * DMM + h * DKK + d0;

    float4 kreg[5], vreg[5];

    auto stage_load = [&](int it) {
        const float* kp = kg + stbase + (size_t)(it * 32) * DMM;
        const float* vp = vg + stbase + (size_t)(it * 32) * DMM;
#pragma unroll
        for (int i = 0; i < 5; ++i) {
            kreg[i] = *(const float4*)(kp + 4 * i);
            vreg[i] = *(const float4*)(vp + 4 * i);
        }
    };
    auto stage_store = [&](int bsel) {
#pragma unroll
        for (int i = 0; i < 5; ++i) {
            union { _Float16 hx[4]; uint2 u2; } pk;
            pk.hx[0] = (_Float16)kreg[i].x; pk.hx[1] = (_Float16)kreg[i].y;
            pk.hx[2] = (_Float16)kreg[i].z; pk.hx[3] = (_Float16)kreg[i].w;
            *(uint2*)&kbuf[bsel][keyl][d0 + 4 * i] = pk.u2;
            vbufT[bsel][d0 + 4 * i + 0][keyl] = (_Float16)vreg[i].x;
            vbufT[bsel][d0 + 4 * i + 1][keyl] = (_Float16)vreg[i].y;
            vbufT[bsel][d0 + 4 * i + 2][keyl] = (_Float16)vreg[i].z;
            vbufT[bsel][d0 + 4 * i + 3][keyl] = (_Float16)vreg[i].w;
        }
        if (qtr == 3) {     // zero-pad d = 80..95
            uint2 z; z.x = 0u; z.y = 0u;
#pragma unroll
            for (int i = 0; i < 4; ++i) *(uint2*)&kbuf[bsel][keyl][80 + 4 * i] = z;
        }
    };

    // ---- Q fragments (A-layout), f32 -> f16, zero pad d >= 80 ----
    Frag16 aq[3];
    {
        const float* qrow = qg + ((size_t)(b * LL + q0 + n)) * DMM + h * DKK;
#pragma unroll
        for (int c = 0; c < 3; ++c) {
#pragma unroll
            for (int j = 0; j < 8; ++j) {
                int kk = 32 * c + kkoff(j, hi);
                _Float16 lo = (_Float16)0.f, hh = (_Float16)0.f;
                if (kk < DKK) { lo = (_Float16)qrow[kk]; hh = (_Float16)qrow[kk + 1]; }
                aq[c].v[2 * j]     = lo;
                aq[c].v[2 * j + 1] = hh;
            }
        }
    }

    v8f oacc[5];
#pragma unroll
    for (int c = 0; c < 5; ++c)
#pragma unroll
        for (int r = 0; r < 8; ++r) oacc[c][r] = 0.f;

    float mrow[8], lrow[8];
#pragma unroll
    for (int r = 0; r < 8; ++r) { mrow[r] = -3.0e38f; lrow[r] = 0.f; }

    // prologue: stage tile 0 into buffer 0
    stage_load(0);
    stage_store(0);

    for (int it = 0; it < LL / 32; ++it) {
        __syncthreads();                 // buffer (it&1) ready; other buffer free
        const int buf  = it & 1;
        const int key0 = it * 32;

        if (it + 1 < LL / 32) stage_load(it + 1);   // overlap with compute

        // ---- scores: s = (2*QK^T - |k|^2)/13  (|q|^2 cancels in softmax) ----
        v8f sfr[2];
#pragma unroll
        for (int sub = 0; sub < 2; ++sub) {
            v8f s;
#pragma unroll
            for (int r = 0; r < 8; ++r) s[r] = 0.f;
#pragma unroll
            for (int c = 0; c < 3; ++c) {
                Frag16 bk;
                const _Float16* src = &kbuf[buf][sub * 16 + n][32 * c + 8 * hi];
                bk.q4[0] = *(const uint4*)(src);
                bk.q4[1] = *(const uint4*)(src + 16);
                s = __builtin_amdgcn_wmma_f32_16x16x32_f16(
                        false, aq[c].v, false, bk.v, (short)0, s, false, false);
            }
            float skv = skdiv[((size_t)(b * HH + h)) * LL + key0 + sub * 16 + n];
#pragma unroll
            for (int r = 0; r < 8; ++r) s[r] = s[r] * (2.0f / 13.0f) - skv;
            sfr[sub] = s;
        }

        // ---- online softmax (row r+8*hi owned by this half-wave) ----
        float alpha[8];
#pragma unroll
        for (int r = 0; r < 8; ++r) {
            float m2 = fmaxf(sfr[0][r], sfr[1][r]);
#pragma unroll
            for (int d = 1; d < 16; d <<= 1) m2 = fmaxf(m2, __shfl_xor(m2, d, 32));
            float mn = fmaxf(mrow[r], m2);
            alpha[r] = __expf(mrow[r] - mn);
            mrow[r]  = mn;
        }
#pragma unroll
        for (int c = 0; c < 5; ++c)
#pragma unroll
            for (int r = 0; r < 8; ++r) oacc[c][r] *= alpha[r];

        float ps[8];
#pragma unroll
        for (int r = 0; r < 8; ++r) ps[r] = 0.f;
#pragma unroll
        for (int sub = 0; sub < 2; ++sub)
#pragma unroll
            for (int r = 0; r < 8; ++r) {
                float p = __expf(sfr[sub][r] - mrow[r]);
                ps[r] += p;
                pbuf[wv][r + 8 * hi][sub * 16 + n] = (_Float16)p;
            }
#pragma unroll
        for (int r = 0; r < 8; ++r) {
            float t = ps[r];
#pragma unroll
            for (int d = 1; d < 16; d <<= 1) t += __shfl_xor(t, d, 32);
            lrow[r] = lrow[r] * alpha[r] + t;
        }

        // wave-private LDS transpose: wait for our DS stores before A-loads
        asm volatile("s_wait_dscnt 0" ::: "memory");

        Frag16 pa;
#pragma unroll
        for (int j = 0; j < 8; ++j) {
            int kk = kkoff(j, hi);
            pa.u[j] = *(const unsigned int*)&pbuf[wv][n][kk];
        }

        // ---- PV: 5 d-chunks of 16, K = 32 keys ----
#pragma unroll
        for (int c = 0; c < 5; ++c) {
            Frag16 bv;
            const _Float16* src = &vbufT[buf][16 * c + n][8 * hi];
            bv.q4[0] = *(const uint4*)(src);
            bv.q4[1] = *(const uint4*)(src + 16);
            oacc[c] = __builtin_amdgcn_wmma_f32_16x16x32_f16(
                          false, pa.v, false, bv.v, (short)0, oacc[c], false, false);
        }

        // stage next tile into the other buffer (safe: all waves that could
        // still be reading it are pre-barrier of this iteration)
        if (it + 1 < LL / 32) stage_store((it + 1) & 1);
    }

    // ---- epilogue: normalize and store f16 attention output ----
#pragma unroll
    for (int c = 0; c < 5; ++c)
#pragma unroll
        for (int r = 0; r < 8; ++r) {
            float val = oacc[c][r] / lrow[r];
            size_t row = (size_t)(b * LL + q0 + r + 8 * hi);
            attn16[row * DMM + h * DKK + 16 * c + n] = (_Float16)val;
        }
}

// ---------------------------------------------------------------------------
// Kernel 2: FC GEMM  out[8192,640] = attn16[8192,640] @ w16[640,640]^T + bias
// Wave computes a 16x64 tile; 20 K-steps of 32.
// ---------------------------------------------------------------------------
__global__ __launch_bounds__(128) void l2mha_fc_kernel(
    const _Float16* __restrict__ a16, const _Float16* __restrict__ w16,
    const float* __restrict__ bias, float* __restrict__ out)
{
    const int tid  = threadIdx.x;
    const int wv   = tid >> 5;
    const int lane = tid & 31;
    const int n    = lane & 15;
    const int hi   = lane >> 4;

    const int bid  = blockIdx.x;
    const int colg = bid % 10;
    const int rt   = (bid / 10) * 4 + wv;
    const int n0   = rt * 16;
    const int m0   = colg * 64;

    v8f acc[4];
#pragma unroll
    for (int c = 0; c < 4; ++c)
#pragma unroll
        for (int r = 0; r < 8; ++r) acc[c][r] = 0.f;

    for (int kc = 0; kc < 20; ++kc) {
        Frag16 a;
        const _Float16* arow = a16 + (size_t)(n0 + n) * DMM + 32 * kc + 8 * hi;
        a.q4[0] = *(const uint4*)(arow);
        a.q4[1] = *(const uint4*)(arow + 16);
#pragma unroll
        for (int c = 0; c < 4; ++c) {
            Frag16 bw;
            const _Float16* wrow = w16 + (size_t)(m0 + 16 * c + n) * DMM + 32 * kc + 8 * hi;
            bw.q4[0] = *(const uint4*)(wrow);
            bw.q4[1] = *(const uint4*)(wrow + 16);
            acc[c] = __builtin_amdgcn_wmma_f32_16x16x32_f16(
                         false, a.v, false, bw.v, (short)0, acc[c], false, false);
        }
    }
#pragma unroll
    for (int c = 0; c < 4; ++c) {
        int m = m0 + 16 * c + n;
        float bi = bias[m];
#pragma unroll
        for (int r = 0; r < 8; ++r)
            out[(size_t)(n0 + r + 8 * hi) * DMM + m] = acc[c][r] + bi;
    }
}

// ---------------------------------------------------------------------------
// Kernel 3: x = fcout*gamma + residual ; y = LayerNorm(x)*ln_w + ln_b
// One 256-thread block per row of 640.
// ---------------------------------------------------------------------------
__global__ __launch_bounds__(256) void l2mha_ln_kernel(
    const float* __restrict__ fcout, const float* __restrict__ residual,
    const float* __restrict__ gamma, const float* __restrict__ lnw,
    const float* __restrict__ lnb, float* __restrict__ out)
{
    __shared__ float sA[256];
    __shared__ float sB[256];
    const int row = blockIdx.x;
    const int tid = threadIdx.x;

    float xv[3];
    float s1 = 0.f, s2 = 0.f;
#pragma unroll
    for (int i = 0; i < 3; ++i) {
        int c = tid + i * 256;
        float x = 0.f;
        if (c < DMM) {
            x = fcout[(size_t)row * DMM + c] * gamma[c] + residual[(size_t)row * DMM + c];
            s1 += x; s2 += x * x;
        }
        xv[i] = x;
    }
    sA[tid] = s1; sB[tid] = s2;
    __syncthreads();
#pragma unroll
    for (int st = 128; st > 0; st >>= 1) {
        if (tid < st) { sA[tid] += sA[tid + st]; sB[tid] += sB[tid + st]; }
        __syncthreads();
    }
    float mu  = sA[0] * (1.0f / DMM);
    float var = sB[0] * (1.0f / DMM) - mu * mu;
    float inv = rsqrtf(var + 1e-5f);
#pragma unroll
    for (int i = 0; i < 3; ++i) {
        int c = tid + i * 256;
        if (c < DMM)
            out[(size_t)row * DMM + c] = (xv[i] - mu) * inv * lnw[c] + lnb[c];
    }
}

// ---------------------------------------------------------------------------
extern "C" void kernel_launch(void* const* d_in, const int* in_sizes, int n_in,
                              void* d_out, int out_size, void* d_ws, size_t ws_size,
                              hipStream_t stream) {
    const float* q     = (const float*)d_in[0];
    const float* k     = (const float*)d_in[1];
    const float* v     = (const float*)d_in[2];
    const float* fc_w  = (const float*)d_in[3];
    const float* fc_b  = (const float*)d_in[4];
    const float* gamma = (const float*)d_in[5];
    const float* ln_w  = (const float*)d_in[6];
    const float* ln_b  = (const float*)d_in[7];

    char* ws = (char*)d_ws;
    float*    skdiv = (float*)ws;                                   // 256 KB
    _Float16* w16   = (_Float16*)(ws + 262144);                     // 800 KB
    _Float16* a16   = (_Float16*)(ws + 262144 + 819200);            // 10 MB
    float*    fcout = (float*)(ws + 262144 + 819200 + 10485760);    // 20 MB

    l2mha_sk_kernel<<<(BB * HH * LL + 255) / 256, 256, 0, stream>>>(k, skdiv);
    l2mha_cvt_kernel<<<(DMM * DMM + 255) / 256, 256, 0, stream>>>(fc_w, w16, DMM * DMM);
    l2mha_attn_kernel<<<BB * HH * (LL / 64), 128, 0, stream>>>(q, k, v, skdiv, a16);
    l2mha_fc_kernel<<<(BB * LL / 64) * (DMM / 64), 128, 0, stream>>>(a16, w16, fc_b, fcout);
    l2mha_ln_kernel<<<BB * LL, 256, 0, stream>>>(fcout, q, gamma, ln_w, ln_b, (float*)d_out);
}
